// sparse_conv_67594195304882
// MI455X (gfx1250) — matrix operand order repaired and checked by reading, not compiled, not run on hardware
//
#include <hip/hip_runtime.h>

// ---------------------------------------------------------------------------
// Sparse block bottleneck (SBNet-style) for MI455X / gfx1250, wave32.
// Matmuls: v_wmma_f32_16x16x32_bf16. Async LDS fill: global_load_async_to_lds.
// ---------------------------------------------------------------------------

#define N_IMG   8
#define C0      64
#define C1      128
#define H_IMG   256
#define W_IMG   256
#define HW      (H_IMG * W_IMG)          // 65536
#define NHW     (N_IMG * HW)             // 524288
#define NB      1024                     // active blocks
#define BS      16
#define PXB     (BS * BS)                // 256 pixels per block
#define NPX     (NB * PXB)               // 262144 gathered pixels
#define EPSBN   1e-5f

typedef __attribute__((ext_vector_type(16))) __bf16 v16bf;
typedef __attribute__((ext_vector_type(8)))  __bf16 v8bf;
typedef __attribute__((ext_vector_type(8)))  float  v8f;

struct US16 { unsigned short u[16]; };

__device__ __forceinline__ unsigned short f2bf(float f) {
    unsigned u = __float_as_uint(f);
    u += 0x7fffu + ((u >> 16) & 1u);     // round-to-nearest-even
    return (unsigned short)(u >> 16);
}
__device__ __forceinline__ float bf2f(unsigned short b) {
    return __uint_as_float(((unsigned)b) << 16);
}
__device__ __forceinline__ v8bf ld8(const unsigned short* p) {
    return *reinterpret_cast<const v8bf*>(p);
}
__device__ __forceinline__ v16bf join8(v8bf lo, v8bf hi) {
    v16bf r;
#pragma unroll
    for (int i = 0; i < 8; ++i) { r[i] = lo[i]; r[i + 8] = hi[i]; }
    return r;
}
__device__ __forceinline__ v8f wmma_bf16(v16bf a, v16bf b, v8f c) {
    return __builtin_amdgcn_wmma_f32_16x16x32_bf16(false, a, false, b,
                                                   (short)0, c, false, false);
}

// --------------------------------------------------------------------------
// 0) zero the stats region (2048 floats) -- must run every launch
// --------------------------------------------------------------------------
__global__ void k_init_stats(float* st) {
    int t = threadIdx.x;
#pragma unroll
    for (int i = 0; i < 8; ++i) st[t + i * 256] = 0.0f;
}

// --------------------------------------------------------------------------
// 1) convert weights to bf16; wd2 transposed to [tap][co][ci]
// --------------------------------------------------------------------------
__global__ void k_cvt_weights(const float* w1, const float* wd1,
                              const float* wd2, const float* wd3,
                              unsigned short* w1b, unsigned short* wd1b,
                              unsigned short* wd2b, unsigned short* wd3b) {
    int i = blockIdx.x * 256 + threadIdx.x;
    if (i < C0 * C0)  w1b[i]  = f2bf(w1[i]);
    if (i < C1 * C0)  wd1b[i] = f2bf(wd1[i]);
    if (i < C0 * C1)  wd3b[i] = f2bf(wd3[i]);
    if (i < 9 * C1 * C1) {
        int ci  = i & 127;
        int co  = (i >> 7) & 127;
        int tap = i >> 14;
        wd2b[i] = f2bf(wd2[(co * C1 + ci) * 9 + tap]);
    }
}

// --------------------------------------------------------------------------
// 2) full-image 1x1 conv (64->64) + bias + relu, per-channel sum/sumsq.
//    One WG per (n, h); 256 pixels = one row. GEMM M=co, N=px, K=ci.
// --------------------------------------------------------------------------
__global__ void __launch_bounds__(256)
k_conv1(const float* __restrict__ x, const unsigned short* __restrict__ w1b,
        const float* __restrict__ b1, float* __restrict__ out,
        float* __restrict__ ssum, float* __restrict__ ssq) {
    __shared__ __align__(16) unsigned short xs[PXB * C0];  // [px][ci] 32KB
    __shared__ __align__(16) unsigned short wsm[C0 * C0];  // [co][ci]  8KB
    __shared__ float lsum[C0], lsq[C0];

    const int t = threadIdx.x;
    const int h = blockIdx.x, n = blockIdx.y;
    const float* xn = x + (size_t)n * C0 * HW + (size_t)h * W_IMG;

    if (t < C0) { lsum[t] = 0.0f; lsq[t] = 0.0f; }
    // transpose-stage input row: float4 over px, scatter bf16 into [px][ci]
    {
        const int cisub = t >> 6, px0 = (t & 63) * 4;
#pragma unroll 4
        for (int it = 0; it < 16; ++it) {
            int ci = it * 4 + cisub;
            float4 v = *reinterpret_cast<const float4*>(xn + (size_t)ci * HW + px0);
            xs[(px0 + 0) * C0 + ci] = f2bf(v.x);
            xs[(px0 + 1) * C0 + ci] = f2bf(v.y);
            xs[(px0 + 2) * C0 + ci] = f2bf(v.z);
            xs[(px0 + 3) * C0 + ci] = f2bf(v.w);
        }
    }
    {
        const uint4* s = reinterpret_cast<const uint4*>(w1b);
        uint4* d = reinterpret_cast<uint4*>(wsm);
        d[t] = s[t]; d[t + 256] = s[t + 256];
    }
    __syncthreads();

    const int wv = t >> 5, l = t & 31, lm = l & 15, kh = l >> 4;

#pragma unroll
    for (int pti = 0; pti < 2; ++pti) {
        const int pt = wv * 2 + pti;
        const int px = pt * 16 + lm;
        v16bf Bf[2];
#pragma unroll
        for (int k = 0; k < 2; ++k) {
            int base = px * C0 + k * 32 + kh * 16;
            Bf[k] = join8(ld8(xs + base), ld8(xs + base + 8));
        }
#pragma unroll
        for (int mt = 0; mt < 4; ++mt) {
            v8f c = {};
#pragma unroll
            for (int k = 0; k < 2; ++k) {
                int ab = (mt * 16 + lm) * C0 + k * 32 + kh * 16;
                v16bf Af = join8(ld8(wsm + ab), ld8(wsm + ab + 8));
                c = wmma_bf16(Af, Bf[k], c);
            }
#pragma unroll
            for (int r = 0; r < 8; ++r) {
                int co = mt * 16 + kh * 8 + r;
                float v = c[r] + b1[co];
                v = v > 0.0f ? v : 0.0f;
                out[((size_t)n * C0 + co) * HW + (size_t)h * W_IMG + px] = v;
                atomicAdd(&lsum[co], v);
                atomicAdd(&lsq[co], v * v);
            }
        }
    }
    __syncthreads();
    if (t < C0) { atomicAdd(&ssum[t], lsum[t]); atomicAdd(&ssq[t], lsq[t]); }
}

// --------------------------------------------------------------------------
// 3) finalize BN: scale = gamma*rsqrt(var+eps), shift = beta - mean*scale
// --------------------------------------------------------------------------
__global__ void k_bnfin(const float* sum, const float* sq,
                        const float* gamma, const float* beta,
                        float* scale, float* shift, int C, float invc) {
    int c = threadIdx.x;
    if (c < C) {
        float m  = sum[c] * invc;
        float vv = sq[c] * invc - m * m;
        float rs = rsqrtf(vv + EPSBN);
        float s  = rs * gamma[c];
        scale[c] = s;
        shift[c] = beta[c] - m * s;
    }
}

// --------------------------------------------------------------------------
// 4) normalize stage-1 output in place (float4 grid-stride)
// --------------------------------------------------------------------------
__global__ void k_norm1(float* out, const float* scale, const float* shift) {
    size_t idx = (size_t)blockIdx.x * 256 + threadIdx.x;
    float4* o4 = reinterpret_cast<float4*>(out);
    for (size_t i = idx; i < (size_t)C0 * NHW / 4; i += (size_t)8192 * 256) {
        int c = (int)((i >> 14) & 63);   // (i*4 >> 16) & 63
        float sc = scale[c], sh = shift[c];
        float4 v = o4[i];
        v.x = v.x * sc + sh; v.y = v.y * sc + sh;
        v.z = v.z * sc + sh; v.w = v.w * sc + sh;
        o4[i] = v;
    }
}

// --------------------------------------------------------------------------
// 5) gather active blocks -> g0 bf16 [block][px][ci]
// --------------------------------------------------------------------------
__global__ void __launch_bounds__(256)
k_gather(const float* __restrict__ out, const int* __restrict__ aidx,
         unsigned short* __restrict__ g0) {
    const int b = blockIdx.x, t = threadIdx.x;
    const int n = aidx[b * 3], h0 = aidx[b * 3 + 1] * BS, w0 = aidx[b * 3 + 2] * BS;
    const int y = t >> 4, xq = t & 15;
#pragma unroll 4
    for (int ci = 0; ci < C0; ++ci) {
        float v = out[((size_t)n * C0 + ci) * HW + (size_t)(h0 + y) * W_IMG + w0 + xq];
        g0[((size_t)b * PXB + t) * C0 + ci] = f2bf(v);
    }
}

// --------------------------------------------------------------------------
// 6) dense 1x1 (64->128) per block + bias + relu + stats, store gA bf16.
//    Activation tile fetched with ASYNCcnt-tracked global->LDS DMA.
// --------------------------------------------------------------------------
__global__ void __launch_bounds__(256)
k_dense1(const unsigned short* __restrict__ g0,
         const unsigned short* __restrict__ wd1b, const float* __restrict__ bd1,
         unsigned short* __restrict__ gA,
         float* __restrict__ ssum, float* __restrict__ ssq) {
    __shared__ __align__(16) unsigned short bsm[PXB * C0];  // 32KB
    __shared__ __align__(16) unsigned short asm_[C1 * C0];  // 16KB
    __shared__ float lsum[C1], lsq[C1];
    const int b = blockIdx.x, t = threadIdx.x;
    if (t < C1) { lsum[t] = 0.0f; lsq[t] = 0.0f; }
    {
        // async DMA of the 32KB activation tile into LDS (ASYNCcnt-tracked)
        const unsigned short* src = g0 + (size_t)b * PXB * C0;
        unsigned lbase = (unsigned)(size_t)(void*)bsm;  // LDS offset = flat[31:0]
#pragma unroll
        for (int i = 0; i < 8; ++i) {
            unsigned voff  = (unsigned)((t + i * 256) * 16);
            unsigned laddr = lbase + voff;
            asm volatile("global_load_async_to_lds_b128 %0, %1, %2"
                         :: "v"(laddr), "v"(voff), "s"(src) : "memory");
        }
        const uint4* sa = reinterpret_cast<const uint4*>(wd1b);
        uint4* da = reinterpret_cast<uint4*>(asm_);
#pragma unroll
        for (int i = 0; i < 4; ++i) da[t + i * 256] = sa[t + i * 256];
        asm volatile("s_wait_asynccnt 0x0" ::: "memory");
    }
    __syncthreads();

    const int wv = t >> 5, l = t & 31, lm = l & 15, kh = l >> 4;
#pragma unroll
    for (int pti = 0; pti < 2; ++pti) {
        const int pt = wv * 2 + pti, px = pt * 16 + lm;
        v16bf Bf[2];
#pragma unroll
        for (int k = 0; k < 2; ++k) {
            int base = px * C0 + k * 32 + kh * 16;
            Bf[k] = join8(ld8(bsm + base), ld8(bsm + base + 8));
        }
#pragma unroll
        for (int mt = 0; mt < 8; ++mt) {
            v8f c = {};
#pragma unroll
            for (int k = 0; k < 2; ++k) {
                int ab = (mt * 16 + lm) * C0 + k * 32 + kh * 16;
                v16bf Af = join8(ld8(asm_ + ab), ld8(asm_ + ab + 8));
                c = wmma_bf16(Af, Bf[k], c);
            }
            US16 pk;
#pragma unroll
            for (int r = 0; r < 8; ++r) {
                int co = mt * 16 + kh * 8 + r;
                float v = c[r] + bd1[co];
                v = v > 0.0f ? v : 0.0f;
                pk.u[r] = f2bf(v); pk.u[r + 8] = 0;
                atomicAdd(&lsum[co], v);
                atomicAdd(&lsq[co], v * v);
            }
            *reinterpret_cast<uint4*>(gA + ((size_t)b * PXB + px) * C1 + mt * 16 + kh * 8) =
                *reinterpret_cast<const uint4*>(pk.u);
        }
    }
    __syncthreads();
    if (t < C1) { atomicAdd(&ssum[t], lsum[t]); atomicAdd(&ssq[t], lsq[t]); }
}

// --------------------------------------------------------------------------
// 7) dense 3x3 (128->128), pad 1, per block. BN(d1) applied on load.
//    Padded 18x18 LDS tile (two 64-ch phases) + per-tap LDS weight staging.
//    grid = (NB, 2): y selects rows 0..7 / 8..15 (one px-tile per wave).
// --------------------------------------------------------------------------
__global__ void __launch_bounds__(256)
k_dense2(const unsigned short* __restrict__ gA,
         const unsigned short* __restrict__ wd2b, const float* __restrict__ bd2,
         const float* __restrict__ sc1, const float* __restrict__ sh1,
         unsigned short* __restrict__ gB,
         float* __restrict__ ssum, float* __restrict__ ssq) {
    __shared__ __align__(16) unsigned short pad[18 * 18 * C0];  // 40.5KB
    __shared__ __align__(16) unsigned short wtap[C1 * C0];      // 16KB
    __shared__ float scs[C1], shs[C1];
    __shared__ float lsum[C1], lsq[C1];
    const int b = blockIdx.x, t = threadIdx.x;
    if (t < C1) { scs[t] = sc1[t]; shs[t] = sh1[t]; lsum[t] = 0.0f; lsq[t] = 0.0f; }

    const int wv = t >> 5, l = t & 31, lm = l & 15, kh = l >> 4;
    const int pt = blockIdx.y * 8 + wv;            // row (0..15)
    v8f acc[8] = {};

    const uint4* wsrc = reinterpret_cast<const uint4*>(wd2b);
    uint4* wdst = reinterpret_cast<uint4*>(wtap);
    const int co2 = t >> 1, hf = t & 1;

    for (int ph = 0; ph < 2; ++ph) {
        __syncthreads();
        // zero full padded tile (18*18*64 = 20736 shorts, 81 per thread)
#pragma unroll
        for (int i = 0; i < 81; ++i) pad[t + i * 256] = 0;
        __syncthreads();
        // interior fill with BN(d1) affine applied
        {
            const int y = t >> 4, xq = t & 15;
            const unsigned short* src = gA + ((size_t)b * PXB + t) * C1 + ph * C0;
#pragma unroll 4
            for (int j = 0; j < C0; ++j) {
                int ci = ph * C0 + j;
                float v = bf2f(src[j]) * scs[ci] + shs[ci];
                pad[((y + 1) * 18 + (xq + 1)) * C0 + j] = f2bf(v);
            }
        }

        for (int tap = 0; tap < 9; ++tap) {
            __syncthreads();           // pad ready (tap 0) / prior tap done
            // stage this tap's [128co x 64ci] weight slice into LDS
#pragma unroll
            for (int i = 0; i < 4; ++i)
                wdst[co2 * 8 + hf * 4 + i] =
                    wsrc[(tap * C1 + co2) * 16 + ph * 8 + hf * 4 + i];
            if (tap < 8)
                __builtin_prefetch(wd2b + (size_t)(tap + 1) * C1 * C1 + ph * C0, 0, 0);
            __syncthreads();

            const int dy = tap / 3, dx = tap % 3;
            const int pp = (pt + dy) * 18 + (lm + dx);
#pragma unroll
            for (int cs = 0; cs < 2; ++cs) {
                int bb = pp * C0 + cs * 32 + kh * 16;
                v16bf Bf = join8(ld8(pad + bb), ld8(pad + bb + 8));
#pragma unroll
                for (int mt = 0; mt < 8; ++mt) {
                    int ab = (mt * 16 + lm) * C0 + cs * 32 + kh * 16;
                    v16bf Af = join8(ld8(wtap + ab), ld8(wtap + ab + 8));
                    acc[mt] = wmma_bf16(Af, Bf, acc[mt]);
                }
            }
        }
    }
    __syncthreads();

    const int px = pt * 16 + lm;
#pragma unroll
    for (int mt = 0; mt < 8; ++mt) {
        US16 pk;
#pragma unroll
        for (int r = 0; r < 8; ++r) {
            int co = mt * 16 + kh * 8 + r;
            float v = acc[mt][r] + bd2[co];
            v = v > 0.0f ? v : 0.0f;
            pk.u[r] = f2bf(v); pk.u[r + 8] = 0;
            atomicAdd(&lsum[co], v);
            atomicAdd(&lsq[co], v * v);
        }
        *reinterpret_cast<uint4*>(gB + ((size_t)b * PXB + px) * C1 + mt * 16 + kh * 8) =
            *reinterpret_cast<const uint4*>(pk.u);
    }
    __syncthreads();
    if (t < C1) { atomicAdd(&ssum[t], lsum[t]); atomicAdd(&ssq[t], lsq[t]); }
}

// --------------------------------------------------------------------------
// 8) dense 1x1 (128->64) per block. BN(d2) applied on fragment load from HBM.
// --------------------------------------------------------------------------
__global__ void __launch_bounds__(256)
k_dense3(const unsigned short* __restrict__ gB,
         const unsigned short* __restrict__ wd3b, const float* __restrict__ bd3,
         const float* __restrict__ sc2, const float* __restrict__ sh2,
         unsigned short* __restrict__ gC,
         float* __restrict__ ssum, float* __restrict__ ssq) {
    __shared__ __align__(16) unsigned short asm_[C0 * C1];  // 16KB
    __shared__ float scs[C1], shs[C1];
    __shared__ float lsum[C0], lsq[C0];
    const int b = blockIdx.x, t = threadIdx.x;
    if (t < C1) { scs[t] = sc2[t]; shs[t] = sh2[t]; }
    if (t < C0) { lsum[t] = 0.0f; lsq[t] = 0.0f; }
    {
        const uint4* sa = reinterpret_cast<const uint4*>(wd3b);
        uint4* da = reinterpret_cast<uint4*>(asm_);
#pragma unroll
        for (int i = 0; i < 4; ++i) da[t + i * 256] = sa[t + i * 256];
    }
    __syncthreads();

    const int wv = t >> 5, l = t & 31, lm = l & 15, kh = l >> 4;
#pragma unroll
    for (int pti = 0; pti < 2; ++pti) {
        const int pt = wv * 2 + pti, px = pt * 16 + lm;
        v8f acc[4] = {};
#pragma unroll
        for (int cs = 0; cs < 4; ++cs) {
            // normalized B fragment straight from global
            US16 raw;
            const unsigned short* src = gB + ((size_t)b * PXB + px) * C1 + cs * 32 + kh * 16;
            *reinterpret_cast<uint4*>(raw.u)     = *reinterpret_cast<const uint4*>(src);
            *reinterpret_cast<uint4*>(raw.u + 8) = *reinterpret_cast<const uint4*>(src + 8);
            US16 nrm;
#pragma unroll
            for (int j = 0; j < 16; ++j) {
                int ci = cs * 32 + kh * 16 + j;
                nrm.u[j] = f2bf(bf2f(raw.u[j]) * scs[ci] + shs[ci]);
            }
            v16bf Bf = __builtin_bit_cast(v16bf, nrm);
#pragma unroll
            for (int mt = 0; mt < 4; ++mt) {
                int ab = (mt * 16 + lm) * C1 + cs * 32 + kh * 16;
                v16bf Af = join8(ld8(asm_ + ab), ld8(asm_ + ab + 8));
                acc[mt] = wmma_bf16(Af, Bf, acc[mt]);
            }
        }
#pragma unroll
        for (int mt = 0; mt < 4; ++mt) {
            US16 pk;
#pragma unroll
            for (int r = 0; r < 8; ++r) {
                int co = mt * 16 + kh * 8 + r;
                float v = acc[mt][r] + bd3[co];
                v = v > 0.0f ? v : 0.0f;
                pk.u[r] = f2bf(v); pk.u[r + 8] = 0;
                atomicAdd(&lsum[co], v);
                atomicAdd(&lsq[co], v * v);
            }
            *reinterpret_cast<uint4*>(gC + ((size_t)b * PXB + px) * C0 + mt * 16 + kh * 8) =
                *reinterpret_cast<const uint4*>(pk.u);
        }
    }
    __syncthreads();
    if (t < C0) { atomicAdd(&ssum[t], lsum[t]); atomicAdd(&ssq[t], lsq[t]); }
}

// --------------------------------------------------------------------------
// 9) scatter: BN(d3) affine + overwrite active blocks in d_out
// --------------------------------------------------------------------------
__global__ void __launch_bounds__(256)
k_scatter(const unsigned short* __restrict__ gC, const int* __restrict__ aidx,
          const float* __restrict__ sc3, const float* __restrict__ sh3,
          float* __restrict__ out) {
    __shared__ float scs[C0], shs[C0];
    const int b = blockIdx.x, t = threadIdx.x;
    if (t < C0) { scs[t] = sc3[t]; shs[t] = sh3[t]; }
    __syncthreads();
    const int n = aidx[b * 3], h0 = aidx[b * 3 + 1] * BS, w0 = aidx[b * 3 + 2] * BS;
    const int y = t >> 4, xq = t & 15;
#pragma unroll 4
    for (int co = 0; co < C0; ++co) {
        float v = bf2f(gC[((size_t)b * PXB + t) * C0 + co]) * scs[co] + shs[co];
        out[((size_t)n * C0 + co) * HW + (size_t)(h0 + y) * W_IMG + w0 + xq] = v;
    }
}

// --------------------------------------------------------------------------
extern "C" void kernel_launch(void* const* d_in, const int* in_sizes, int n_in,
                              void* d_out, int out_size, void* d_ws, size_t ws_size,
                              hipStream_t stream) {
    const float* x    = (const float*)d_in[0];
    const int*   aidx = (const int*)  d_in[1];
    const float* w1   = (const float*)d_in[2];
    const float* b1   = (const float*)d_in[3];
    const float* g1   = (const float*)d_in[4];
    const float* be1  = (const float*)d_in[5];
    const float* wd1  = (const float*)d_in[6];
    const float* bd1  = (const float*)d_in[7];
    const float* gd1  = (const float*)d_in[8];
    const float* bed1 = (const float*)d_in[9];
    const float* wd2  = (const float*)d_in[10];
    const float* bd2  = (const float*)d_in[11];
    const float* gd2  = (const float*)d_in[12];
    const float* bed2 = (const float*)d_in[13];
    const float* wd3  = (const float*)d_in[14];
    const float* bd3  = (const float*)d_in[15];
    const float* gd3  = (const float*)d_in[16];
    const float* bed3 = (const float*)d_in[17];
    float* out = (float*)d_out;

    // workspace layout
    float* st = (float*)d_ws;                                 // 2048 f32 (8KB)
    // sums: s1[0/64] sd1[128/256] sd2[384/512] sd3[640/704]
    // affine: f1[768/832] fd1[896/1024] fd2[1152/1280] fd3[1408/1472]
    unsigned short* w1b  = (unsigned short*)((char*)d_ws + 8192);
    unsigned short* wd1b = w1b  + C0 * C0;                    // 4096
    unsigned short* wd2b = wd1b + C1 * C0;                    // 8192
    unsigned short* wd3b = wd2b + 9 * C1 * C1;                // 147456
    unsigned short* g0 = (unsigned short*)((char*)d_ws + (size_t)(1u << 20));
    unsigned short* gA = (unsigned short*)((char*)d_ws + (size_t)(1u << 20) + ((size_t)32 << 20));
    unsigned short* gB = (unsigned short*)((char*)d_ws + (size_t)(1u << 20) + ((size_t)96 << 20));
    unsigned short* gC = g0;                                  // alias (g0 dead)

    k_init_stats<<<1, 256, 0, stream>>>(st);
    k_cvt_weights<<<576, 256, 0, stream>>>(w1, wd1, wd2, wd3, w1b, wd1b, wd2b, wd3b);

    k_conv1<<<dim3(H_IMG, N_IMG), 256, 0, stream>>>(x, w1b, b1, out, st + 0, st + 64);
    k_bnfin<<<1, 128, 0, stream>>>(st + 0, st + 64, g1, be1, st + 768, st + 832,
                                   C0, 1.0f / (float)NHW);
    k_norm1<<<8192, 256, 0, stream>>>(out, st + 768, st + 832);
    k_gather<<<NB, 256, 0, stream>>>(out, aidx, g0);

    k_dense1<<<NB, 256, 0, stream>>>(g0, wd1b, bd1, gA, st + 128, st + 256);
    k_bnfin<<<1, 128, 0, stream>>>(st + 128, st + 256, gd1, bed1, st + 896, st + 1024,
                                   C1, 1.0f / (float)NPX);

    k_dense2<<<dim3(NB, 2), 256, 0, stream>>>(gA, wd2b, bd2, st + 896, st + 1024,
                                              gB, st + 384, st + 512);
    k_bnfin<<<1, 128, 0, stream>>>(st + 384, st + 512, gd2, bed2, st + 1152, st + 1280,
                                   C1, 1.0f / (float)NPX);

    k_dense3<<<NB, 256, 0, stream>>>(gB, wd3b, bd3, st + 1152, st + 1280,
                                     gC, st + 640, st + 704);
    k_bnfin<<<1, 128, 0, stream>>>(st + 640, st + 704, gd3, bed3, st + 1408, st + 1472,
                                   C0, 1.0f / (float)NPX);

    k_scatter<<<NB, 256, 0, stream>>>(gC, aidx, st + 1408, st + 1472, out);
}
